// RobertaEncoderLayer_88124138979553
// MI455X (gfx1250) — matrix operand order repaired and checked by reading, not compiled, not run on hardware
//
#include <hip/hip_runtime.h>

// ---------------------------------------------------------------------------
// DeBERTa-style disentangled-attention encoder layer for gfx1250 (MI455X).
// All GEMMs via v_wmma_f32_16x16x32_bf16 (wave32 WMMA, fp32 accum), with
// double-buffered GLOBAL_LOAD_ASYNC_TO_LDS_B128 tile staging (ASYNCcnt).
// ---------------------------------------------------------------------------

typedef __attribute__((ext_vector_type(16))) __bf16 v16bf;
typedef __attribute__((ext_vector_type(8)))  float  v8f;
typedef unsigned short u16;

union Frag16 { v16bf v; unsigned int q[8]; };

__device__ __forceinline__ u16 f2bf(float f) {
  unsigned int u = __float_as_uint(f);
  return (u16)((u + 0x7fffu + ((u >> 16) & 1u)) >> 16);  // round-to-nearest-even
}

// A-fragment: 16x32 bf16, per CDNA5 ISA 7.12.2 (16-bit A-matrix layout).
__device__ __forceinline__ v16bf afrag(const u16* lA, int m, int hl) {
  Frag16 f;
  const u16* row = lA + m * 32;
#pragma unroll
  for (int v = 0; v < 8; ++v) {
    int k = (v < 4) ? (8 * hl + 2 * v) : (16 + 8 * hl + 2 * (v - 4));
    f.q[v] = *(const unsigned int*)(row + k);
  }
  return f.v;
}

// B-fragment: 32x16 bf16. lB stored [n][k] ([64][32]).
__device__ __forceinline__ v16bf bfrag(const u16* lB, int n, int hl) {
  Frag16 f;
  const u16* row = lB + n * 32 + 8 * hl;
#pragma unroll
  for (int v = 0; v < 8; ++v) f.q[v] = *(const unsigned int*)(row + 2 * v + ((v < 4) ? 0 : 8));
  return f.v;
}

__device__ __forceinline__ v8f wmma_bf16(v16bf a, v16bf b, v8f c) {
  return __builtin_amdgcn_wmma_f32_16x16x32_bf16(false, a, false, b, (short)0, c,
                                                 false, false);
}

__device__ __forceinline__ unsigned lds_off(const void* p) {
  // flat encoding of an LDS address keeps the byte offset in the low 32 bits
  return (unsigned)(size_t)p;
}

// ---------------------------------------------------------------------------
// Generic batched WMMA GEMM: C[b,h] = act(A[b,h] @ B[b,h](^T) + bias) (+C)(+res)
// Block tile 128x64, BK=32; 8 waves (4x2), each wave 32x32 (4 wmma / K-step).
// Double-buffered LDS; global->LDS via global_load_async_to_lds_b128.
// Requires M%128==0, N%64==0, K%32==0 (true for all launches below).
// ---------------------------------------------------------------------------
template <bool TB, bool RELU, bool OBF16, bool ACC, bool BIAS, bool RES>
__global__ __launch_bounds__(256) void gemm_wmma(
    const u16* __restrict__ A, const u16* __restrict__ Bm,
    const float* __restrict__ bias, const float* __restrict__ res,
    void* __restrict__ Cout, int K, int lda, int ldb, int ldc,
    long sAb, long sAh, long sBb, long sBh, long sCb, long sCh, int H) {
  __shared__ u16 lA[2][128 * 32];   // [buf][m][k]
  __shared__ u16 lB[2][64 * 32];    // [buf][n][k]
  const int t = threadIdx.x;
  const int lane = t & 31;
  const int wv = t >> 5;
  const int wm = wv >> 1, wn = wv & 1;
  const int gm0 = blockIdx.y * 128, gn0 = blockIdx.x * 64;
  const int z = blockIdx.z;
  const int b = z / H, h = z % H;
  const u16* Ab = A + (long)b * sAb + (long)h * sAh;
  const u16* Bb = Bm + (long)b * sBb + (long)h * sBh;
  const long coff = (long)b * sCb + (long)h * sCh;

  // per-thread staging geometry
  const int arow = t >> 1, acol = (t & 1) << 4;   // A: 32 bytes per thread
  const int bn = t >> 2, bk0 = (t & 3) << 3;      // B (TB): 16 bytes per thread
  const int bk = t & 31, bn0 = (t >> 5) << 3;     // B (non-TB transpose path)

  auto issue = [&](int kk, int buf) {
    {  // A tile 128x32: two async b128 per thread (offset applies to both sides)
      unsigned la = lds_off(&lA[buf][t << 4]);
      unsigned long long ga =
          (unsigned long long)(Ab + (long)(gm0 + arow) * lda + kk + acol);
      asm volatile(
          "global_load_async_to_lds_b128 %0, %1, off\n\t"
          "global_load_async_to_lds_b128 %0, %1, off offset:16"
          :: "v"(la), "v"(ga) : "memory");
    }
    if (TB) {  // B stored [N,K]: rows go straight into lB[n][k]
      unsigned lb = lds_off(&lB[buf][bn * 32 + bk0]);
      unsigned long long gb =
          (unsigned long long)(Bb + (long)(gn0 + bn) * ldb + kk + bk0);
      asm volatile("global_load_async_to_lds_b128 %0, %1, off"
                   :: "v"(lb), "v"(gb) : "memory");
    } else {   // B stored [K,N]: transpose through VGPRs into lB[n][k]
      union { uint4 v; u16 s[8]; } u;
      u.v = *(const uint4*)(Bb + (long)(kk + bk) * ldb + gn0 + bn0);
#pragma unroll
      for (int e = 0; e < 8; ++e) lB[buf][(bn0 + e) * 32 + bk] = u.s[e];
    }
  };

  v8f acc[2][2] = {};
  const int T = K >> 5;

  issue(0, 0);
  for (int i = 0; i < T; ++i) {
    const int buf = i & 1;
    if (i + 1 < T) {
      issue((i + 1) << 5, buf ^ 1);
      // drain only the previous stage's async copies (2 or 3 per wave/stage)
      if (TB) asm volatile("s_wait_asynccnt 3" ::: "memory");
      else    asm volatile("s_wait_asynccnt 2" ::: "memory");
    } else {
      asm volatile("s_wait_asynccnt 0" ::: "memory");
    }
    __syncthreads();  // also waits dscnt/loadcnt for the transpose path

    const u16* lAb = lA[buf];
    const u16* lBb = lB[buf];
    const int hl = lane >> 4;
    const int ml = lane & 15;
    v16bf a0 = afrag(lAb, wm * 32 + ml, hl);
    v16bf a1 = afrag(lAb, wm * 32 + 16 + ml, hl);
    v16bf b0 = bfrag(lBb, wn * 32 + ml, hl);
    v16bf b1 = bfrag(lBb, wn * 32 + 16 + ml, hl);
    acc[0][0] = wmma_bf16(a0, b0, acc[0][0]);
    acc[0][1] = wmma_bf16(a0, b1, acc[0][1]);
    acc[1][0] = wmma_bf16(a1, b0, acc[1][0]);
    acc[1][1] = wmma_bf16(a1, b1, acc[1][1]);
    __syncthreads();  // all waves done reading buf before it is refilled
  }

  // C/D layout: VGPR r, lanes 0-15 -> M=r, lanes 16-31 -> M=r+8; N=lane&15
  const int hl = lane >> 4, nl = lane & 15;
#pragma unroll
  for (int sm = 0; sm < 2; ++sm) {
#pragma unroll
    for (int sn = 0; sn < 2; ++sn) {
      int gnc = gn0 + wn * 32 + sn * 16 + nl;
      float bv = BIAS ? bias[gnc] : 0.0f;
#pragma unroll
      for (int r = 0; r < 8; ++r) {
        int gmr = gm0 + wm * 32 + sm * 16 + hl * 8 + r;
        long ci = coff + (long)gmr * ldc + gnc;
        float val = acc[sm][sn][r] + bv;
        if (RELU) val = fmaxf(val, 0.0f);
        if (ACC) val += ((const float*)Cout)[ci];
        if (RES) val += res[ci];
        if (OBF16) ((u16*)Cout)[ci] = f2bf(val);
        else       ((float*)Cout)[ci] = val;
      }
    }
  }
}

// ---------------------------------------------------------------------------
// LayerNorm over D=1024 per row, fp32 in -> bf16 out.
// ---------------------------------------------------------------------------
__global__ __launch_bounds__(256) void ln_kernel(const float* __restrict__ x,
                                                 const float* __restrict__ g,
                                                 const float* __restrict__ bb,
                                                 u16* __restrict__ y) {
  const int D = 1024;
  const long row = blockIdx.x;
  const int t = threadIdx.x;
  __shared__ float red[256];
  const float* xr = x + row * D;
  float v[4];
  float s = 0.f;
#pragma unroll
  for (int u = 0; u < 4; ++u) { v[u] = xr[t + u * 256]; s += v[u]; }
  red[t] = s; __syncthreads();
  for (int o = 128; o > 0; o >>= 1) { if (t < o) red[t] += red[t + o]; __syncthreads(); }
  float mu = red[0] * (1.f / 1024.f); __syncthreads();
  float s2 = 0.f;
#pragma unroll
  for (int u = 0; u < 4; ++u) { float d = v[u] - mu; s2 += d * d; }
  red[t] = s2; __syncthreads();
  for (int o = 128; o > 0; o >>= 1) { if (t < o) red[t] += red[t + o]; __syncthreads(); }
  float rs = rsqrtf(red[0] * (1.f / 1024.f) + 1e-5f);
#pragma unroll
  for (int u = 0; u < 4; ++u) {
    int c = t + u * 256;
    y[row * D + c] = f2bf((v[u] - mu) * rs * g[c] + bb[c]);
  }
}

// ---------------------------------------------------------------------------
// Fused score assembly + softmax + relative-bin bucketing.
// One block per (b,h,i) row. score = (c2c + qr[i,rp] + kr[j,rp]) * 1/sqrt(3*DH)
// ---------------------------------------------------------------------------
__global__ __launch_bounds__(256) void softmax_kernel(
    const float* __restrict__ c2c, const float* __restrict__ qr,
    const float* __restrict__ kr, const int* __restrict__ rel_pos,
    const unsigned char* __restrict__ mask, u16* __restrict__ attn,
    u16* __restrict__ pr) {
  const int S = 1024, R = 64, H = 16;
  const int i = blockIdx.x, h = blockIdx.y, b = blockIdx.z;
  const int t = threadIdx.x;
  __shared__ float red[256];
  __shared__ float bins[64];
  const long bh = (long)(b * H + h);
  const long rowbase = (bh * S + i) * S;
  const float* qrow = qr + (bh * S + i) * R;
  const float* krb = kr + bh * S * R;
  const int* rprow = rel_pos + ((long)b * S + i) * S;
  const float scale = 0.07216878364870323f;  // 1/sqrt(3*64)
  float sc[4], e4[4];
  int rp4[4];
  float mx = -3.4e38f;
#pragma unroll
  for (int u = 0; u < 4; ++u) {
    int j = t + u * 256;
    int rij = rprow[j];
    float s = (c2c[rowbase + j] + qrow[rij] + krb[(long)j * R + rij]) * scale;
    if (!mask[(long)b * S + j]) s = -1e9f;
    sc[u] = s; rp4[u] = rij;
    mx = fmaxf(mx, s);
  }
  red[t] = mx; __syncthreads();
  for (int o = 128; o > 0; o >>= 1) { if (t < o) red[t] = fmaxf(red[t], red[t + o]); __syncthreads(); }
  mx = red[0]; __syncthreads();
  float sum = 0.f;
#pragma unroll
  for (int u = 0; u < 4; ++u) { e4[u] = __expf(sc[u] - mx); sum += e4[u]; }
  red[t] = sum; __syncthreads();
  for (int o = 128; o > 0; o >>= 1) { if (t < o) red[t] += red[t + o]; __syncthreads(); }
  float inv = 1.f / red[0];
  if (t < 64) bins[t] = 0.f;
  __syncthreads();
#pragma unroll
  for (int u = 0; u < 4; ++u) {
    float a = e4[u] * inv;
    attn[rowbase + t + u * 256] = f2bf(a);
    atomicAdd(&bins[rp4[u]], a);   // ds_add_f32 bucketing into R bins
  }
  __syncthreads();
  if (t < 64) pr[(bh * S + i) * R + t] = f2bf(bins[t]);
}

__global__ __launch_bounds__(256) void cvt_kernel(const float* __restrict__ in,
                                                  u16* __restrict__ out, long n) {
  long i = (long)blockIdx.x * 256 + threadIdx.x;
  if (i < n) out[i] = f2bf(in[i]);
}

// ---------------------------------------------------------------------------
extern "C" void kernel_launch(void* const* d_in, const int* in_sizes, int n_in,
                              void* d_out, int out_size, void* d_ws, size_t ws_size,
                              hipStream_t stream) {
  (void)in_sizes; (void)n_in; (void)out_size; (void)ws_size;
  const int B = 2, S = 1024, H = 16, D = 1024, R = 64, FF = 4096, DH = 64;
  const long MD = (long)B * S * D;          // 2M
  const long BHSS = (long)B * H * S * S;    // 32M
  const long BHSR = (long)B * H * S * R;    // 2M

  const float* src  = (const float*)d_in[0];
  const unsigned char* mask = (const unsigned char*)d_in[1];
  const int* rel_pos = (const int*)d_in[2];
  const float* rel_q = (const float*)d_in[3];
  const float* rel_k = (const float*)d_in[4];
  const float* rel_v = (const float*)d_in[5];
  const float* Wq = (const float*)d_in[6];  const float* bq = (const float*)d_in[7];
  const float* Wk = (const float*)d_in[8];  const float* bk = (const float*)d_in[9];
  const float* Wv = (const float*)d_in[10]; const float* bv = (const float*)d_in[11];
  const float* Wo = (const float*)d_in[12]; const float* bo = (const float*)d_in[13];
  const float* ln1g = (const float*)d_in[14]; const float* ln1b = (const float*)d_in[15];
  const float* W1 = (const float*)d_in[16]; const float* b1 = (const float*)d_in[17];
  const float* W2 = (const float*)d_in[18]; const float* b2 = (const float*)d_in[19];
  const float* ln2g = (const float*)d_in[20]; const float* ln2b = (const float*)d_in[21];

  // ---- workspace carving ----
  char* base = (char*)d_ws;
  size_t off = 0;
  auto carve = [&](size_t bytes) -> void* {
    void* p = base + off;
    off = (off + bytes + 255) & ~(size_t)255;
    return p;
  };
  u16* xnorm = (u16*)carve(MD * 2);
  u16* qb    = (u16*)carve(MD * 2);
  u16* kb    = (u16*)carve(MD * 2);
  u16* vb    = (u16*)carve(MD * 2);
  u16* Wqb   = (u16*)carve((size_t)D * D * 2);
  u16* Wkb   = (u16*)carve((size_t)D * D * 2);
  u16* Wvb   = (u16*)carve((size_t)D * D * 2);
  u16* Wob   = (u16*)carve((size_t)D * D * 2);
  u16* W1b   = (u16*)carve((size_t)D * FF * 2);
  u16* W2b   = (u16*)carve((size_t)FF * D * 2);
  u16* rqb   = (u16*)carve((size_t)R * D * 2);
  u16* rkb   = (u16*)carve((size_t)R * D * 2);
  u16* rvb   = (u16*)carve((size_t)R * D * 2);
  float* qrf = (float*)carve(BHSR * 4);
  float* krf = (float*)carve(BHSR * 4);
  float* c2c = (float*)carve(BHSS * 4);
  u16* attn  = (u16*)carve(BHSS * 2);
  u16* pr    = (u16*)carve(BHSR * 2);
  float* ctx = (float*)carve(MD * 4);
  u16* ctxb  = (u16*)carve(MD * 2);
  float* xbuf = (float*)carve(MD * 4);
  u16* y2    = (u16*)carve(MD * 2);
  u16* f1    = (u16*)carve((size_t)B * S * FF * 2);

  auto cvt = [&](const float* in, u16* out, long n) {
    cvt_kernel<<<dim3((unsigned)((n + 255) / 256)), 256, 0, stream>>>(in, out, n);
  };
  cvt(Wq, Wqb, (long)D * D);
  cvt(Wk, Wkb, (long)D * D);
  cvt(Wv, Wvb, (long)D * D);
  cvt(Wo, Wob, (long)D * D);
  cvt(W1, W1b, (long)D * FF);
  cvt(W2, W2b, (long)FF * D);
  cvt(rel_q, rqb, (long)R * D);
  cvt(rel_k, rkb, (long)R * D);
  cvt(rel_v, rvb, (long)R * D);

  // LN1 -> bf16
  ln_kernel<<<dim3(B * S), 256, 0, stream>>>(src, ln1g, ln1b, xnorm);

  // Q/K/V projections (M=2048, N=1024, K=1024), bf16 out
  gemm_wmma<false, false, true, false, true, false>
      <<<dim3(D / 64, (B * S) / 128, 1), 256, 0, stream>>>(
          xnorm, Wqb, bq, nullptr, qb, D, D, D, D, 0, 0, 0, 0, 0, 0, 1);
  gemm_wmma<false, false, true, false, true, false>
      <<<dim3(D / 64, (B * S) / 128, 1), 256, 0, stream>>>(
          xnorm, Wkb, bk, nullptr, kb, D, D, D, D, 0, 0, 0, 0, 0, 0, 1);
  gemm_wmma<false, false, true, false, true, false>
      <<<dim3(D / 64, (B * S) / 128, 1), 256, 0, stream>>>(
          xnorm, Wvb, bv, nullptr, vb, D, D, D, D, 0, 0, 0, 0, 0, 0, 1);

  // c2c[b,h] = q_bh @ k_bh^T   (M=1024, N=1024, K=64), fp32 out
  gemm_wmma<true, false, false, false, false, false>
      <<<dim3(S / 64, S / 128, B * H), 256, 0, stream>>>(
          qb, kb, nullptr, nullptr, c2c, DH, D, D, S,
          (long)S * D, DH, (long)S * D, DH, (long)H * S * S, (long)S * S, H);

  // qr[b,h] = q_bh @ rel_k_h^T  (M=1024, N=64, K=64)
  gemm_wmma<true, false, false, false, false, false>
      <<<dim3(R / 64, S / 128, B * H), 256, 0, stream>>>(
          qb, rkb, nullptr, nullptr, qrf, DH, D, D, R,
          (long)S * D, DH, 0, DH, (long)H * S * R, (long)S * R, H);
  // kr[b,h] = k_bh @ rel_q_h^T
  gemm_wmma<true, false, false, false, false, false>
      <<<dim3(R / 64, S / 128, B * H), 256, 0, stream>>>(
          kb, rqb, nullptr, nullptr, krf, DH, D, D, R,
          (long)S * D, DH, 0, DH, (long)H * S * R, (long)S * R, H);

  // softmax + relative-bin bucketing
  softmax_kernel<<<dim3(S, H, B), 256, 0, stream>>>(c2c, qrf, krf, rel_pos, mask,
                                                    attn, pr);

  // ctx = attn @ v  (M=1024, N=64, K=1024), fp32
  gemm_wmma<false, false, false, false, false, false>
      <<<dim3(DH / 64, S / 128, B * H), 256, 0, stream>>>(
          attn, vb, nullptr, nullptr, ctx, S, S, D, D,
          (long)H * S * S, (long)S * S, (long)S * D, DH, (long)S * D, DH, H);
  // ctx += pr @ rel_v  (M=1024, N=64, K=64), accumulate
  gemm_wmma<false, false, false, true, false, false>
      <<<dim3(DH / 64, S / 128, B * H), 256, 0, stream>>>(
          pr, rvb, nullptr, nullptr, ctx, R, R, D, D,
          (long)H * S * R, (long)S * R, 0, DH, (long)S * D, DH, H);

  cvt(ctx, ctxb, MD);

  // x = src + ctx @ Wo + bo  (fp32)
  gemm_wmma<false, false, false, false, true, true>
      <<<dim3(D / 64, (B * S) / 128, 1), 256, 0, stream>>>(
          ctxb, Wob, bo, src, xbuf, D, D, D, D, 0, 0, 0, 0, 0, 0, 1);

  // LN2 -> bf16
  ln_kernel<<<dim3(B * S), 256, 0, stream>>>(xbuf, ln2g, ln2b, y2);

  // f1 = relu(y @ W1 + b1)  (M=2048, N=4096, K=1024), bf16 out
  gemm_wmma<false, true, true, false, true, false>
      <<<dim3(FF / 64, (B * S) / 128, 1), 256, 0, stream>>>(
          y2, W1b, b1, nullptr, f1, D, D, FF, FF, 0, 0, 0, 0, 0, 0, 1);

  // out = x + f1 @ W2 + b2  (M=2048, N=1024, K=4096), fp32 -> d_out
  gemm_wmma<false, false, false, false, true, true>
      <<<dim3(D / 64, (B * S) / 128, 1), 256, 0, stream>>>(
          f1, W2b, b2, xbuf, (float*)d_out, FF, FF, D, D, 0, 0, 0, 0, 0, 0, 1);
}
